// HybridPointerNetwork_14164802143060
// MI455X (gfx1250) — matrix operand order repaired and checked by reading, not compiled, not run on hardware
//
#include <hip/hip_runtime.h>
#include <hip/hip_bf16.h>
#include <math.h>

typedef __attribute__((ext_vector_type(16))) _Float16 v16h;
typedef __attribute__((ext_vector_type(8)))  _Float16 v8h;
typedef __attribute__((ext_vector_type(4)))  _Float16 v4h;
typedef __attribute__((ext_vector_type(8)))  float    v8f;
typedef __attribute__((ext_vector_type(4)))  float    v4f;

#define N_NODES 128
#define BATCH   128
#define HID     512
#define NHEAD   8
#define HD      64
#define TLEN    129     // N + 1
#define FFDIM   2048
#define LDPAD   144     // padded score row stride for decoder (multiple of 16)

#define GF_TRANSB 1
#define GF_RELU   2
#define GF_BIAS   4

// ---------------------------------------------------------------------------
// Generic batched GEMM: C = A(MxK) * B + bias, optional relu.
// B is (KxN) row-major if !TRANSB, (NxK) row-major if TRANSB.
// fp32 in/out, f16 WMMA (v_wmma_f32_16x16x32_f16) internally.
// Block = 128 threads (4 waves). Block tile 64x64, wave tile 32x32, K-step 32.
// Interior tiles: vectorized v4f global loads + ds_store_b64.
// Edge tiles: clamped addresses + mask-multiply (keeps loads unconditional,
// prevents LLVM from sinking loads into a divergent waterfall).
// ---------------------------------------------------------------------------
__global__ __launch_bounds__(128)
void wmma_gemm_kernel(const float* __restrict__ A, const float* __restrict__ Bm,
                      const float* __restrict__ bias, float* __restrict__ C,
                      int M, int Nn, int K, int lda, int ldb, int ldc,
                      long long offA1, long long offA2,
                      long long offB1, long long offB2,
                      long long offC1, long long offC2,
                      int batch2, int flags)
{
    __shared__ _Float16 Ah[64][32];   // A tile, row-major [m][k]
    __shared__ _Float16 Bt[64][32];   // B tile, transposed [n][k]

    const int z  = blockIdx.z;
    const int z1 = z / batch2;
    const int z2 = z - z1 * batch2;
    A  += (long long)z1 * offA1 + (long long)z2 * offA2;
    Bm += (long long)z1 * offB1 + (long long)z2 * offB2;
    C  += (long long)z1 * offC1 + (long long)z2 * offC2;

    const int m0   = blockIdx.y * 64;
    const int n0   = blockIdx.x * 64;
    const int tid  = threadIdx.x;
    const int lane = tid & 31;
    const int w    = tid >> 5;
    const int wm   = (w >> 1) * 32;
    const int wn   = (w & 1) * 32;
    const int lrow = lane & 15;
    const int koffA  = (lane < 16) ? 0 : 8;    // A: half-wave splits K in 8s
    const int kbaseB = (lane < 16) ? 0 : 16;   // unused: B fragments also use koffA (8s)
    const bool transB = (flags & GF_TRANSB) != 0;
    const bool fullM  = (m0 + 64 <= M);
    const bool fullN  = (n0 + 64 <= Nn);

    v8f acc[2][2] = {};

    for (int k0 = 0; k0 < K; k0 += 32) {
        const bool fullK = (k0 + 32 <= K);

        // ---- stage A (fp32 -> f16) ----
        if (fullM && fullK) {
            // 512 v4f loads, 4 per thread; row = f>>3, kq = (f&7)*4
            #pragma unroll
            for (int i = 0; i < 4; ++i) {
                int f = tid + i * 128;
                int r = f >> 3;
                int kq = (f & 7) << 2;
                v4f v = *(const v4f*)(A + (long long)(m0 + r) * lda + (k0 + kq));
                v4h h;
                h[0] = (_Float16)v[0]; h[1] = (_Float16)v[1];
                h[2] = (_Float16)v[2]; h[3] = (_Float16)v[3];
                *(v4h*)&Ah[r][kq] = h;
            }
        } else {
            // clamped + mask-multiply, fully unrolled: loads stay unconditional
            #pragma unroll
            for (int i = 0; i < 16; ++i) {
                int e = tid + i * 128;
                int r = e >> 5, k = e & 31;
                int gm = m0 + r, gk = k0 + k;
                int cm = gm < M ? gm : M - 1;
                int ck = gk < K ? gk : K - 1;
                float v = A[(long long)cm * lda + ck];
                float msk = (gm < M && gk < K) ? 1.f : 0.f;
                Ah[r][k] = (_Float16)(v * msk);
            }
        }

        // ---- stage B transposed [n][k] ----
        if (transB) {
            if (fullN && fullK) {
                #pragma unroll
                for (int i = 0; i < 4; ++i) {
                    int f = tid + i * 128;
                    int r = f >> 3;
                    int kq = (f & 7) << 2;
                    v4f v = *(const v4f*)(Bm + (long long)(n0 + r) * ldb + (k0 + kq));
                    v4h h;
                    h[0] = (_Float16)v[0]; h[1] = (_Float16)v[1];
                    h[2] = (_Float16)v[2]; h[3] = (_Float16)v[3];
                    *(v4h*)&Bt[r][kq] = h;
                }
            } else {
                #pragma unroll
                for (int i = 0; i < 16; ++i) {
                    int e = tid + i * 128;
                    int n = e >> 5, k = e & 31;
                    int gn = n0 + n, gk = k0 + k;
                    int cn = gn < Nn ? gn : Nn - 1;
                    int ck = gk < K ? gk : K - 1;
                    float v = Bm[(long long)cn * ldb + ck];
                    float msk = (gn < Nn && gk < K) ? 1.f : 0.f;
                    Bt[n][k] = (_Float16)(v * msk);
                }
            }
        } else {
            if (fullN && fullK) {
                // load contiguous along n, scatter-transpose into LDS
                #pragma unroll
                for (int i = 0; i < 4; ++i) {
                    int f = tid + i * 128;
                    int k = f >> 4;            // 0..31
                    int nq = (f & 15) << 2;    // 0..60
                    v4f v = *(const v4f*)(Bm + (long long)(k0 + k) * ldb + (n0 + nq));
                    Bt[nq + 0][k] = (_Float16)v[0];
                    Bt[nq + 1][k] = (_Float16)v[1];
                    Bt[nq + 2][k] = (_Float16)v[2];
                    Bt[nq + 3][k] = (_Float16)v[3];
                }
            } else {
                #pragma unroll
                for (int i = 0; i < 16; ++i) {
                    int e = tid + i * 128;
                    int n = e >> 5, k = e & 31;
                    int gn = n0 + n, gk = k0 + k;
                    int cn = gn < Nn ? gn : Nn - 1;
                    int ck = gk < K ? gk : K - 1;
                    float v = Bm[(long long)ck * ldb + cn];
                    float msk = (gn < Nn && gk < K) ? 1.f : 0.f;
                    Bt[n][k] = (_Float16)(v * msk);
                }
            }
        }
        __syncthreads();

        v16h afr[2], bfr[2];
        #pragma unroll
        for (int i = 0; i < 2; ++i) {
            int r = wm + i * 16 + lrow;
            v8h lo = *(const v8h*)&Ah[r][koffA];        // K 0..7   (or 8..15)
            v8h hi = *(const v8h*)&Ah[r][16 + koffA];   // K 16..23 (or 24..31)
            #pragma unroll
            for (int q = 0; q < 8; ++q) { afr[i][q] = lo[q]; afr[i][q + 8] = hi[q]; }
        }
        #pragma unroll
        for (int j = 0; j < 2; ++j) {
            int c = wn + j * 16 + lrow;
            v8h lo = *(const v8h*)&Bt[c][koffA];        // K 0..7   (or 8..15)
            v8h hi = *(const v8h*)&Bt[c][16 + koffA];   // K 16..23 (or 24..31)
            #pragma unroll
            for (int q = 0; q < 8; ++q) { bfr[j][q] = lo[q]; bfr[j][q + 8] = hi[q]; }
        }
        #pragma unroll
        for (int i = 0; i < 2; ++i)
            #pragma unroll
            for (int j = 0; j < 2; ++j)
                acc[i][j] = __builtin_amdgcn_wmma_f32_16x16x32_f16(
                    false, afr[i], false, bfr[j], (short)0, acc[i][j], false, false);
        __syncthreads();
    }

    // C/D layout: lane<16 -> rows 0..7, lane>=16 -> rows 8..15; col = lane&15
    const int rbase = (lane < 16) ? 0 : 8;
    #pragma unroll
    for (int i = 0; i < 2; ++i) {
        #pragma unroll
        for (int j = 0; j < 2; ++j) {
            int col = n0 + wn + j * 16 + lrow;
            if (col < Nn) {
                float bv = (flags & GF_BIAS) ? bias[col] : 0.f;
                #pragma unroll
                for (int r = 0; r < 8; ++r) {
                    int row = m0 + wm + i * 16 + rbase + r;
                    if (row < M) {
                        float v = acc[i][j][r] + bv;
                        if (flags & GF_RELU) v = fmaxf(v, 0.f);
                        C[(long long)row * ldc + col] = v;
                    }
                }
            }
        }
    }
}

// ---------------------------------------------------------------------------
// out_row = LN(x_row + res_row) * g + b     (H = 512, 256 threads, 2 elem/thr)
// ---------------------------------------------------------------------------
__global__ __launch_bounds__(256)
void ln_residual_kernel(const float* __restrict__ x, const float* __restrict__ res,
                        const float* __restrict__ g, const float* __restrict__ bb,
                        float* __restrict__ out, int rpb_in, int rpb_out)
{
    __shared__ float red[256];
    const int row = blockIdx.x;
    const int bI  = row / rpb_in;
    const int r   = row - bI * rpb_in;
    const float* xp = x   + (long long)row * HID;
    const float* rp = res + (long long)row * HID;
    float* op = out + ((long long)bI * rpb_out + r) * HID;
    const int tid = threadIdx.x;

    float v0 = xp[tid] + rp[tid];
    float v1 = xp[tid + 256] + rp[tid + 256];

    red[tid] = v0 + v1; __syncthreads();
    for (int s = 128; s > 0; s >>= 1) { if (tid < s) red[tid] += red[tid + s]; __syncthreads(); }
    float mean = red[0] * (1.f / HID); __syncthreads();

    float d0 = v0 - mean, d1 = v1 - mean;
    red[tid] = d0 * d0 + d1 * d1; __syncthreads();
    for (int s = 128; s > 0; s >>= 1) { if (tid < s) red[tid] += red[tid + s]; __syncthreads(); }
    float rstd = rsqrtf(red[0] * (1.f / HID) + 1e-5f);

    op[tid]       = d0 * rstd * g[tid]       + bb[tid];
    op[tid + 256] = d1 * rstd * g[tid + 256] + bb[tid + 256];
}

// ---------------------------------------------------------------------------
// In-place softmax over score rows: p = softmax(scale * s[0..valid)),
// zeros written to [valid, ld) so downstream GEMM pad reads are clean.
// ---------------------------------------------------------------------------
__global__ __launch_bounds__(128)
void attn_softmax_kernel(float* __restrict__ S, int Lq, int Lk, int ld,
                         int causal, float scale)
{
    __shared__ float red[128];
    const int row = blockIdx.x;
    const long long z = blockIdx.y;
    float* p = S + (z * (long long)Lq + row) * (long long)ld;
    const int tid = threadIdx.x;
    const int valid = causal ? (row + 1) : Lk;

    float m = -1e30f;
    for (int c = tid; c < valid; c += 128) m = fmaxf(m, p[c] * scale);
    red[tid] = m; __syncthreads();
    for (int s = 64; s > 0; s >>= 1) { if (tid < s) red[tid] = fmaxf(red[tid], red[tid + s]); __syncthreads(); }
    m = red[0]; __syncthreads();

    float sum = 0.f;
    for (int c = tid; c < valid; c += 128) { float e = expf(p[c] * scale - m); p[c] = e; sum += e; }
    red[tid] = sum; __syncthreads();
    for (int s = 64; s > 0; s >>= 1) { if (tid < s) red[tid] += red[tid + s]; __syncthreads(); }
    float inv = 1.f / red[0];

    for (int c = tid; c < valid; c += 128) p[c] *= inv;
    for (int c = valid + tid; c < ld; c += 128) p[c] = 0.f;
}

// dec_seq[b,0] = dec_start; dec_seq[b,t] = ext_feats[b, target[b,t-1]]
__global__ __launch_bounds__(128)
void build_dec_seq_kernel(const float* __restrict__ enc_input, const int* __restrict__ target,
                          const float* __restrict__ dec_start, float* __restrict__ dec_seq)
{
    int t = blockIdx.x, b = blockIdx.y, tid = threadIdx.x;
    float* dst = dec_seq + ((long long)b * TLEN + t) * HID;
    if (t == 0) { for (int i = tid; i < HID; i += 128) dst[i] = dec_start[i]; return; }
    int idx = target[b * TLEN + (t - 1)];
    if (idx >= N_NODES) { for (int i = tid; i < HID; i += 128) dst[i] = 0.f; return; }
    const float* src = enc_input + ((long long)b * N_NODES + idx) * HID;
    for (int i = tid; i < HID; i += 128) dst[i] = src[i];
}

__global__ __launch_bounds__(128)
void fill_eos_kernel(const float* __restrict__ eos, float* __restrict__ ext_enc)
{
    int b = blockIdx.x, tid = threadIdx.x;
    float* dst = ext_enc + ((long long)b * TLEN + N_NODES) * HID;
    for (int i = tid; i < HID; i += 128) dst[i] = eos[i];
}

// masked log-softmax NLL; target rows are permutations of [0, T)
__global__ __launch_bounds__(256)
void pointer_loss_kernel(const float* __restrict__ logits, const int* __restrict__ target,
                         float* __restrict__ out)
{
    __shared__ int   pos[TLEN];
    __shared__ float red[256];
    int b = blockIdx.x, tid = threadIdx.x;
    for (int s = tid; s < TLEN; s += 256) pos[target[b * TLEN + s]] = s;  // inverse perm
    __syncthreads();

    float acc = 0.f;
    for (int t = tid; t < TLEN; t += 256) {
        const float* row = logits + ((long long)b * TLEN + t) * TLEN;
        int tgt = target[b * TLEN + t];
        float m = -1e30f;
        for (int n = 0; n < TLEN; ++n) if (pos[n] >= t) m = fmaxf(m, row[n]);
        float s = 0.f;
        for (int n = 0; n < TLEN; ++n) if (pos[n] >= t) s += expf(row[n] - m);
        acc += -(row[tgt] - m - logf(s));
    }
    red[tid] = acc; __syncthreads();
    for (int s = 128; s > 0; s >>= 1) { if (tid < s) red[tid] += red[tid + s]; __syncthreads(); }
    if (tid == 0) atomicAdd(out, red[0] / (float)(BATCH * TLEN));
}

__global__ void zero_kernel(float* p) { p[0] = 0.f; }

// ---------------------------------------------------------------------------
static inline void launch_gemm(const float* A, const float* B, const float* bias, float* C,
                               int M, int N, int K, int lda, int ldb, int ldc,
                               long long oA1, long long oA2, long long oB1, long long oB2,
                               long long oC1, long long oC2,
                               int batch, int batch2, int flags, hipStream_t stream)
{
    dim3 g((N + 63) / 64, (M + 63) / 64, batch);
    wmma_gemm_kernel<<<g, dim3(128), 0, stream>>>(A, B, bias, C, M, N, K, lda, ldb, ldc,
                                                  oA1, oA2, oB1, oB2, oC1, oC2, batch2, flags);
}

extern "C" void kernel_launch(void* const* d_in, const int* in_sizes, int n_in,
                              void* d_out, int out_size, void* d_ws, size_t ws_size,
                              hipStream_t stream)
{
    const float* adj       = (const float*)d_in[0];
    const int*   target    = (const int*)  d_in[1];
    const float* W_embed   = (const float*)d_in[2];
    const float* b_embed   = (const float*)d_in[3];
    const float* dec_start = (const float*)d_in[4];
    const float* enc_eos   = (const float*)d_in[5];
    const float* enc_qkv_w = (const float*)d_in[6];
    const float* enc_qkv_b = (const float*)d_in[7];
    const float* enc_out_w = (const float*)d_in[8];
    const float* enc_out_b = (const float*)d_in[9];
    const float* enc_ln1_g = (const float*)d_in[10];
    const float* enc_ln1_b = (const float*)d_in[11];
    const float* enc_ff1_w = (const float*)d_in[12];
    const float* enc_ff1_b = (const float*)d_in[13];
    const float* enc_ff2_w = (const float*)d_in[14];
    const float* enc_ff2_b = (const float*)d_in[15];
    const float* enc_ln2_g = (const float*)d_in[16];
    const float* enc_ln2_b = (const float*)d_in[17];
    const float* dec_qkv_w = (const float*)d_in[18];
    const float* dec_qkv_b = (const float*)d_in[19];
    const float* dec_out_w = (const float*)d_in[20];
    const float* dec_out_b = (const float*)d_in[21];
    const float* dec_ln1_g = (const float*)d_in[22];
    const float* dec_ln1_b = (const float*)d_in[23];
    const float* dec_ff1_w = (const float*)d_in[24];
    const float* dec_ff1_b = (const float*)d_in[25];
    const float* dec_ff2_w = (const float*)d_in[26];
    const float* dec_ff2_b = (const float*)d_in[27];
    const float* dec_ln2_g = (const float*)d_in[28];
    const float* dec_ln2_b = (const float*)d_in[29];

    float* ws = (float*)d_ws;
    // ---- workspace layout (floats) ----
    const long long ENC_IN  = 0;                                          // B*N*H
    const long long EXT_ENC = ENC_IN  + (long long)BATCH * N_NODES * HID; // B*T*H
    const long long DEC_SEQ = EXT_ENC + (long long)BATCH * TLEN * HID;
    const long long DEC_OUT = DEC_SEQ + (long long)BATCH * TLEN * HID;
    const long long X1      = DEC_OUT + (long long)BATCH * TLEN * HID;
    const long long TMP     = X1      + (long long)BATCH * TLEN * HID;
    const long long QKV     = TMP     + (long long)BATCH * TLEN * HID;    // B*T*3H
    const long long SCORES  = QKV     + (long long)BATCH * TLEN * 3 * HID;// 1024*144*144
    const long long FFB     = QKV;   // FF (B*T*FF) reuses dead QKV+SCORES region
    const long long LOGITS  = QKV;   // logits (B*T*T) reuse after FF dead

    const int  ME = BATCH * N_NODES;   // 16384 encoder rows
    const int  MD = BATCH * TLEN;      // 16512 decoder rows
    const int  BH = BATCH * NHEAD;     // 1024 attention batches
    const float scale = 1.f / 8.f;     // 1/sqrt(64)

    // ================= encoder =================
    // enc_input = adj @ W_embed + b_embed
    launch_gemm(adj, W_embed, b_embed, ws + ENC_IN, ME, HID, N_NODES,
                N_NODES, HID, HID, 0,0, 0,0, 0,0, 1, 1, GF_BIAS, stream);
    // qkv = enc_input @ qkv_w + qkv_b
    launch_gemm(ws + ENC_IN, enc_qkv_w, enc_qkv_b, ws + QKV, ME, 3 * HID, HID,
                HID, 3 * HID, 3 * HID, 0,0, 0,0, 0,0, 1, 1, GF_BIAS, stream);
    // scores[b,h] = q @ k^T      (z = b*NH + h)
    launch_gemm(ws + QKV, ws + QKV + HID, nullptr, ws + SCORES,
                N_NODES, N_NODES, HD, 3 * HID, 3 * HID, N_NODES,
                (long long)N_NODES * 3 * HID, HD,
                (long long)N_NODES * 3 * HID, HD,
                (long long)NHEAD * N_NODES * N_NODES, (long long)N_NODES * N_NODES,
                BH, NHEAD, GF_TRANSB, stream);
    attn_softmax_kernel<<<dim3(N_NODES, BH), dim3(128), 0, stream>>>(
        ws + SCORES, N_NODES, N_NODES, N_NODES, 0, scale);
    // attn_out[b,h] = probs @ v  -> TMP (B,N,H)
    launch_gemm(ws + SCORES, ws + QKV + 2 * HID, nullptr, ws + TMP,
                N_NODES, HD, N_NODES, N_NODES, 3 * HID, HID,
                (long long)NHEAD * N_NODES * N_NODES, (long long)N_NODES * N_NODES,
                (long long)N_NODES * 3 * HID, HD,
                (long long)N_NODES * HID, HD,
                BH, NHEAD, 0, stream);
    // proj = attn_out @ out_w + out_b -> X1
    launch_gemm(ws + TMP, enc_out_w, enc_out_b, ws + X1, ME, HID, HID,
                HID, HID, HID, 0,0, 0,0, 0,0, 1, 1, GF_BIAS, stream);
    // x = LN(enc_input + proj) -> TMP
    ln_residual_kernel<<<dim3(ME), dim3(256), 0, stream>>>(
        ws + ENC_IN, ws + X1, enc_ln1_g, enc_ln1_b, ws + TMP, N_NODES, N_NODES);
    // ff = relu(x @ ff1 + b1) -> FFB ; @ ff2 + b2 -> X1
    launch_gemm(ws + TMP, enc_ff1_w, enc_ff1_b, ws + FFB, ME, FFDIM, HID,
                HID, FFDIM, FFDIM, 0,0, 0,0, 0,0, 1, 1, GF_BIAS | GF_RELU, stream);
    launch_gemm(ws + FFB, enc_ff2_w, enc_ff2_b, ws + X1, ME, HID, FFDIM,
                FFDIM, HID, HID, 0,0, 0,0, 0,0, 1, 1, GF_BIAS, stream);
    // enc_out = LN(x + ff) -> EXT_ENC rows [b][0..N-1] (row-stride T per batch)
    ln_residual_kernel<<<dim3(ME), dim3(256), 0, stream>>>(
        ws + TMP, ws + X1, enc_ln2_g, enc_ln2_b, ws + EXT_ENC, N_NODES, TLEN);
    fill_eos_kernel<<<dim3(BATCH), dim3(128), 0, stream>>>(enc_eos, ws + EXT_ENC);

    // ================= decoder =================
    build_dec_seq_kernel<<<dim3(TLEN, BATCH), dim3(128), 0, stream>>>(
        ws + ENC_IN, target, dec_start, ws + DEC_SEQ);
    launch_gemm(ws + DEC_SEQ, dec_qkv_w, dec_qkv_b, ws + QKV, MD, 3 * HID, HID,
                HID, 3 * HID, 3 * HID, 0,0, 0,0, 0,0, 1, 1, GF_BIAS, stream);
    launch_gemm(ws + QKV, ws + QKV + HID, nullptr, ws + SCORES,
                TLEN, TLEN, HD, 3 * HID, 3 * HID, LDPAD,
                (long long)TLEN * 3 * HID, HD,
                (long long)TLEN * 3 * HID, HD,
                (long long)NHEAD * TLEN * LDPAD, (long long)TLEN * LDPAD,
                BH, NHEAD, GF_TRANSB, stream);
    attn_softmax_kernel<<<dim3(TLEN, BH), dim3(128), 0, stream>>>(
        ws + SCORES, TLEN, TLEN, LDPAD, 1, scale);
    launch_gemm(ws + SCORES, ws + QKV + 2 * HID, nullptr, ws + TMP,
                TLEN, HD, TLEN, LDPAD, 3 * HID, HID,
                (long long)NHEAD * TLEN * LDPAD, (long long)TLEN * LDPAD,
                (long long)TLEN * 3 * HID, HD,
                (long long)TLEN * HID, HD,
                BH, NHEAD, 0, stream);
    launch_gemm(ws + TMP, dec_out_w, dec_out_b, ws + X1, MD, HID, HID,
                HID, HID, HID, 0,0, 0,0, 0,0, 1, 1, GF_BIAS, stream);
    ln_residual_kernel<<<dim3(MD), dim3(256), 0, stream>>>(
        ws + DEC_SEQ, ws + X1, dec_ln1_g, dec_ln1_b, ws + TMP, TLEN, TLEN);
    launch_gemm(ws + TMP, dec_ff1_w, dec_ff1_b, ws + FFB, MD, FFDIM, HID,
                HID, FFDIM, FFDIM, 0,0, 0,0, 0,0, 1, 1, GF_BIAS | GF_RELU, stream);
    launch_gemm(ws + FFB, dec_ff2_w, dec_ff2_b, ws + X1, MD, HID, FFDIM,
                FFDIM, HID, HID, 0,0, 0,0, 0,0, 1, 1, GF_BIAS, stream);
    ln_residual_kernel<<<dim3(MD), dim3(256), 0, stream>>>(
        ws + TMP, ws + X1, dec_ln2_g, dec_ln2_b, ws + DEC_OUT, TLEN, TLEN);

    // logits[b] = dec_out (TxH) @ ext_enc^T (TxH)  (batched NT)
    launch_gemm(ws + DEC_OUT, ws + EXT_ENC, nullptr, ws + LOGITS,
                TLEN, TLEN, HID, HID, HID, TLEN,
                (long long)TLEN * HID, 0,
                (long long)TLEN * HID, 0,
                (long long)TLEN * TLEN, 0,
                BATCH, 1, GF_TRANSB, stream);

    zero_kernel<<<dim3(1), dim3(1), 0, stream>>>((float*)d_out);
    pointer_loss_kernel<<<dim3(BATCH), dim3(256), 0, stream>>>(
        ws + LOGITS, target, (float*)d_out);
}
